// MultiViewImportanceRenderer_49220325212419
// MI455X (gfx1250) — compile-verified, matrix operand order checked by reading, not run on hardware
//
#include <hip/hip_runtime.h>
#include <hip/hip_bf16.h>
#include <math.h>

// ---------------------------------------------------------------------------
// MultiViewImportanceRenderer for MI455X (gfx1250, wave32, WMMA)
// ---------------------------------------------------------------------------

typedef __attribute__((ext_vector_type(16))) _Float16 v16h;
typedef __attribute__((ext_vector_type(8)))  _Float16 v8h;
typedef __attribute__((ext_vector_type(8)))  float    v8f;

#define RAY_START 2.25f
#define RAY_END   3.30f
#define SDEPTH    48
#define NRAYS     8192             // B*R = 2*4096
#define MC        (NRAYS * SDEPTH) // 393216 samples per pass
#define HPLANE    256
#define CCH       32
#define LOG2E     1.4426950408889634f
#define LN2       0.6931471805599453f

struct __attribute__((aligned(4))) f2u { float x, y; };  // 4B-aligned pair load

__device__ __forceinline__ uint32_t pcg_hash(uint32_t v) {
    v = v * 747796405u + 2891336453u;
    v = ((v >> ((v >> 28u) + 4u)) ^ v) * 277803737u;
    return (v >> 22u) ^ v;
}
__device__ __forceinline__ float u01(uint32_t v) {
    return (float)(pcg_hash(v) >> 8) * (1.0f / 16777216.0f);
}
// branchless: max(x,0) + ln2*log2(1+exp2(-|x|*log2e))   (native v_exp/v_log)
__device__ __forceinline__ float softplusf(float x) {
    return fmaxf(x, 0.f) + LN2 * log2f(1.f + exp2f(-fabsf(x) * LOG2E));
}
__device__ __forceinline__ float sigmoidf(float x) {
    return 1.f / (1.f + exp2f(-x * LOG2E));
}
__device__ __forceinline__ float expnegf(float t) {   // exp(-t)
    return exp2f(-t * LOG2E);
}
__device__ __forceinline__ v16h mk16(const _Float16* lo, const _Float16* hi) {
    v8h a = *(const v8h*)lo;
    v8h b = *(const v8h*)hi;
    return __builtin_shufflevector(a, b, 0,1,2,3,4,5,6,7,8,9,10,11,12,13,14,15);
}
// pair-load weights: value = wl*texel[p] + wh*texel[p+1], p = clamp(t,0,254);
// reproduces zero-pad bilinear taps at t and t+1 exactly.
__device__ __forceinline__ void pair_weights(int t, float fr, int& p, float& wl, float& wh) {
    p = min(max(t, 0), HPLANE - 2);
    bool interior = (t >= 0) & (t <= HPLANE - 2);
    bool lo_edge  = (t == -1);
    bool hi_edge  = (t == HPLANE - 1);
    wl = interior ? (1.f - fr) : (lo_edge ? fr : 0.f);
    wh = interior ? fr         : (hi_edge ? (1.f - fr) : 0.f);
}

// ---------------------------------------------------------------------------
// Kernel 1: stratified coarse depths (deterministic hash jitter)
// ---------------------------------------------------------------------------
__global__ __launch_bounds__(256) void gen_depths(float* __restrict__ depthsC) {
    int m = blockIdx.x * 256 + threadIdx.x;
    if (m >= MC) return;
    int i = m % SDEPTH;
    const float delta = (RAY_END - RAY_START) / (float)(SDEPTH - 1);
    float u = u01((uint32_t)m * 2654435761u + 0x9E3779B9u);
    depthsC[m] = RAY_START + ((float)i + u) * delta;
}

// ---------------------------------------------------------------------------
// Kernel 2/4: fused triplane gather + mix + 2-layer MLP via WMMA.
// One wave = one 16-sample tile. 8 waves / block. Grid = MC/128 blocks.
// ---------------------------------------------------------------------------
__global__ __launch_bounds__(256) void fused_sample_mlp(
    const float* __restrict__ front, const float* __restrict__ back,
    const float* __restrict__ ro,    const float* __restrict__ rd,
    const float* __restrict__ wmixg, const float* __restrict__ w1g,
    const float* __restrict__ b1g,   const float* __restrict__ w2g,
    const float* __restrict__ b2g,   const float* __restrict__ depths,
    float* __restrict__ colors_out,  float* __restrict__ sigma_out)
{
    __shared__ __align__(16) _Float16 sW1t[64 * 96];   // [n][k]  12288 B
    __shared__ __align__(16) _Float16 sW2t[48 * 64];   // [n][k]   6144 B
    __shared__ float sB1[64];
    __shared__ float sB2[48];
    __shared__ float sWsig[96];
    __shared__ __align__(16) _Float16 sX[8][16 * 96];  // per-wave features 24576 B
    __shared__ __align__(16) _Float16 sH[8][16 * 64];  // per-wave hidden   16384 B

    const int tid = threadIdx.x;
    // --- stage weights (transposed so B operands are contiguous 16B chunks) ---
    for (int i = tid; i < 64 * 96; i += 256) {
        int n = i / 96, k = i % 96;
        sW1t[i] = (_Float16)w1g[k * 64 + n];
    }
    for (int i = tid; i < 48 * 64; i += 256) {
        int n = i / 64, k = i % 64;
        sW2t[i] = (_Float16)(n < 33 ? w2g[k * 33 + n] : 0.f);
    }
    if (tid < 64) sB1[tid] = b1g[tid];
    if (tid < 48) sB2[tid] = (tid < 33) ? b2g[tid] : 0.f;
    if (tid < 96) sWsig[tid] = sigmoidf(wmixg[tid]);
    __syncthreads();

    const int wv   = tid >> 5;
    const int lane = tid & 31;
    const int srow = lane & 15;       // sample-in-tile (A row / B col)
    const int hi16 = lane >> 4;       // upper half-wave
    const int koff = hi16 * 8;        // A/B 16-bit operand K sub-offset

    const int tile = blockIdx.x * 8 + wv;
    const int m0   = tile * 16;

    // ------------------- featurize 16 samples into sX -------------------
    {
        const int m   = m0 + srow;
        const int ray = m / SDEPTH;
        const int bat = ray >> 12;                     // 4096 rays / batch
        const float d  = depths[m];
        const float ox = ro[ray * 3 + 0], oy = ro[ray * 3 + 1], oz = ro[ray * 3 + 2];
        const float dx = rd[ray * 3 + 0], dy = rd[ray * 3 + 1], dz = rd[ray * 3 + 2];
        const float cx = 2.f * (ox + d * dx);
        const float cy = 2.f * (oy + d * dy);
        const float cz = 2.f * (oz + d * dz);

        #pragma unroll
        for (int p = 0; p < 3; ++p) {
            float gx, gy;
            if (p == 0)      { gx = cx; gy = cy; }
            else if (p == 1) { gx = cx; gy = cz; }
            else             { gx = cz; gy = cx; }
            float fx = (gx + 1.f) * (HPLANE * 0.5f) - 0.5f;
            float fy = (gy + 1.f) * (HPLANE * 0.5f) - 0.5f;
            int   x0 = (int)floorf(fx), y0 = (int)floorf(fy);
            float wx = fx - (float)x0,  wy = fy - (float)y0;

            int xp, yp; float wlx, whx, wly, why;
            pair_weights(x0, wx, xp, wlx, whx);
            pair_weights(y0, wy, yp, wly, why);

            const size_t o00 = (size_t)yp * HPLANE + xp;
            const size_t o10 = o00 + HPLANE;
            const size_t pbase = ((size_t)(bat * 3 + p) * CCH) << 16;  // *65536
            #pragma unroll 4
            for (int cc = 0; cc < 16; ++cc) {
                int ch = hi16 * 16 + cc;
                const float* F  = front + pbase + ((size_t)ch << 16);
                const float* Bk = back  + pbase + ((size_t)ch << 16);
                f2u f0 = *(const f2u*)(F + o00);
                f2u f1 = *(const f2u*)(F + o10);
                f2u b0 = *(const f2u*)(Bk + o00);
                f2u b1 = *(const f2u*)(Bk + o10);
                float vf = wly * (wlx * f0.x + whx * f0.y)
                         + why * (wlx * f1.x + whx * f1.y);
                float vb = wly * (wlx * b0.x + whx * b0.y)
                         + why * (wlx * b1.x + whx * b1.y);
                float wm = sWsig[p * CCH + ch];
                sX[wv][srow * 96 + p * CCH + ch] = (_Float16)(wm * vf + (1.f - wm) * vb);
            }
        }
    }
    __syncthreads();

    // ------------------- GEMM1: X(16x96) * W1(96x64) via WMMA -------------------
    v8f acc[4] = { v8f{}, v8f{}, v8f{}, v8f{} };
    const _Float16* Xw = &sX[wv][0];
    #pragma unroll
    for (int kb = 0; kb < 3; ++kb) {
        const _Float16* arow = Xw + srow * 96 + kb * 32 + koff;
        v16h A = mk16(arow, arow + 16);
        #pragma unroll
        for (int nt = 0; nt < 4; ++nt) {
            const _Float16* brow = sW1t + (nt * 16 + srow) * 96 + kb * 32 + koff;
            v16h Bm = mk16(brow, brow + 16);
            acc[nt] = __builtin_amdgcn_wmma_f32_16x16x32_f16(
                false, A, false, Bm, (short)0, acc[nt], false, false);
        }
    }

    // bias + softplus -> hidden staging (A-layout rows), branchless softplus
    {
        const int mrow = hi16 * 8;
        #pragma unroll
        for (int nt = 0; nt < 4; ++nt) {
            int n = nt * 16 + srow;
            float bn = sB1[n];
            #pragma unroll
            for (int r = 0; r < 8; ++r)
                sH[wv][(mrow + r) * 64 + n] = (_Float16)softplusf(acc[nt][r] + bn);
        }
    }
    __syncthreads();

    // ------------------- GEMM2: H(16x64) * W2(64x48pad) via WMMA -------------------
    v8f o[3] = { v8f{}, v8f{}, v8f{} };
    const _Float16* Hw = &sH[wv][0];
    #pragma unroll
    for (int kb = 0; kb < 2; ++kb) {
        const _Float16* arow = Hw + srow * 64 + kb * 32 + koff;
        v16h A = mk16(arow, arow + 16);
        #pragma unroll
        for (int nt = 0; nt < 3; ++nt) {
            const _Float16* brow = sW2t + (nt * 16 + srow) * 64 + kb * 32 + koff;
            v16h Bm = mk16(brow, brow + 16);
            o[nt] = __builtin_amdgcn_wmma_f32_16x16x32_f16(
                false, A, false, Bm, (short)0, o[nt], false, false);
        }
    }

    // ------------------- epilogue: sigma + sigmoid rgb -------------------
    {
        const int mrow = hi16 * 8;
        #pragma unroll
        for (int nt = 0; nt < 3; ++nt) {
            int n = nt * 16 + srow;
            float bn = sB2[n];
            #pragma unroll
            for (int r = 0; r < 8; ++r) {
                float val = o[nt][r] + bn;
                int mg = m0 + mrow + r;
                if (n == 0) {
                    sigma_out[mg] = val;
                } else if (n <= 32) {
                    colors_out[(size_t)mg * 32 + (n - 1)] =
                        sigmoidf(val) * 1.002f - 0.001f;
                }
            }
        }
    }
}

// ---------------------------------------------------------------------------
// Kernel 3: per-ray coarse march weights -> smoothed PDF -> inverse CDF
// ---------------------------------------------------------------------------
__global__ __launch_bounds__(256) void importance_sample(
    const float* __restrict__ depthsC, const float* __restrict__ sigmaC,
    float* __restrict__ depthsF)
{
    int ray = blockIdx.x * 256 + threadIdx.x;
    if (ray >= NRAYS) return;
    const float* z = depthsC + (size_t)ray * SDEPTH;
    const float* s = sigmaC  + (size_t)ray * SDEPTH;

    float w[SDEPTH - 1];
    float T = 1.f;
    for (int i = 0; i < SDEPTH - 1; ++i) {
        float delta = z[i + 1] - z[i];
        float dens  = softplusf(0.5f * (s[i] + s[i + 1]) - 1.f);
        float alpha = 1.f - expnegf(dens * delta);
        w[i] = alpha * T;
        T *= (1.f - alpha + 1e-10f);
    }
    // max-filter + smooth (wp padded with -inf)
    float wm[SDEPTH];                       // 48
    wm[0] = w[0];
    for (int i = 1; i < SDEPTH - 1; ++i) wm[i] = fmaxf(w[i - 1], w[i]);
    wm[SDEPTH - 1] = w[SDEPTH - 2];
    float wa[SDEPTH - 1];                   // 47
    for (int i = 0; i < SDEPTH - 1; ++i) wa[i] = 0.5f * (wm[i] + wm[i + 1]) + 0.01f;

    // pdf over wa[1..45]  (45 bins), bins = z_mid (47)
    float bins[SDEPTH - 1];
    for (int i = 0; i < SDEPTH - 1; ++i) bins[i] = 0.5f * (z[i] + z[i + 1]);
    const int NS = SDEPTH - 3;              // 45
    float cdf[SDEPTH - 2];                  // 46
    float sum = 0.f;
    for (int j = 0; j < NS; ++j) sum += wa[1 + j] + 1e-5f;
    float inv = 1.f / sum, acc = 0.f;
    cdf[0] = 0.f;
    for (int j = 0; j < NS; ++j) { acc += (wa[1 + j] + 1e-5f) * inv; cdf[j + 1] = acc; }

    float fine[SDEPTH];
    for (int k = 0; k < SDEPTH; ++k) {
        float u = u01((uint32_t)(ray * SDEPTH + k) * 1664525u + 1013904223u);
        int ind = NS + 1;
        for (int j = 0; j <= NS; ++j) { if (cdf[j] > u) { ind = j; break; } }
        int below = max(ind - 1, 0);
        int above = min(ind, NS);
        float c0 = cdf[below], c1 = cdf[above];
        float b0 = bins[below], b1 = bins[above];
        float den = c1 - c0;
        if (den < 1e-5f) den = 1.f;
        fine[k] = b0 + (u - c0) / den * (b1 - b0);
    }
    // insertion sort (48)
    for (int a = 1; a < SDEPTH; ++a) {
        float v = fine[a]; int b2 = a - 1;
        while (b2 >= 0 && fine[b2] > v) { fine[b2 + 1] = fine[b2]; --b2; }
        fine[b2 + 1] = v;
    }
    for (int k = 0; k < SDEPTH; ++k) depthsF[(size_t)ray * SDEPTH + k] = fine[k];
}

// ---------------------------------------------------------------------------
// Kernel 5: merge coarse+fine (both sorted), final 35-channel ray march
// ---------------------------------------------------------------------------
__global__ __launch_bounds__(256) void final_march(
    const float* __restrict__ depthsC, const float* __restrict__ depthsF,
    const float* __restrict__ sigmaC,  const float* __restrict__ sigmaF,
    const float* __restrict__ colorC,  const float* __restrict__ colorF,
    const float* __restrict__ ro,      const float* __restrict__ rd,
    float* __restrict__ out)
{
    int ray = blockIdx.x * 256 + threadIdx.x;
    if (ray >= NRAYS) return;
    const float* dC = depthsC + (size_t)ray * SDEPTH;
    const float* dF = depthsF + (size_t)ray * SDEPTH;
    const float ox = ro[ray * 3 + 0], oy = ro[ray * 3 + 1], oz = ro[ray * 3 + 2];
    const float vx = rd[ray * 3 + 0], vy = rd[ray * 3 + 1], vz = rd[ray * 3 + 2];

    float comp[35];
    for (int k = 0; k < 35; ++k) comp[k] = 0.f;
    float prevc[35], prevs = 0.f, prevd = 0.f;
    float T = 1.f, wsum = 0.f, wd = 0.f;
    float dmin = 0.f, dmax = 0.f;
    int i = 0, j = 0;

    for (int n = 0; n < 2 * SDEPTH; ++n) {
        bool takeC = (i < SDEPTH) && (j >= SDEPTH || dC[i] <= dF[j]);
        int idx; float d; const float* cb; const float* sb;
        if (takeC) { idx = i; d = dC[i]; cb = colorC; sb = sigmaC; ++i; }
        else       { idx = j; d = dF[j]; cb = colorF; sb = sigmaF; ++j; }
        size_t g = (size_t)ray * SDEPTH + idx;
        float cur[35];
        for (int k = 0; k < 32; ++k) cur[k] = cb[g * 32 + k];
        cur[32] = ox + d * vx;
        cur[33] = oy + d * vy;
        cur[34] = oz + d * vz;
        float sg = sb[g];
        if (n == 0) {
            dmin = d;
        } else {
            float delta = d - prevd;
            float dens  = softplusf(0.5f * (sg + prevs) - 1.f);
            float alpha = 1.f - expnegf(dens * delta);
            float wgt   = alpha * T;
            T *= (1.f - alpha + 1e-10f);
            for (int k = 0; k < 35; ++k) comp[k] += wgt * 0.5f * (cur[k] + prevc[k]);
            wsum += wgt;
            wd   += wgt * 0.5f * (d + prevd);
        }
        for (int k = 0; k < 35; ++k) prevc[k] = cur[k];
        prevs = sg; prevd = d; dmax = d;
    }

    float depth = wd / wsum;
    if (!(depth == depth)) depth = INFINITY;    // NaN -> inf
    depth = fminf(fmaxf(depth, dmin), dmax);

    float* out_rgb = out;                       // NRAYS*32
    float* out_dep = out + (size_t)NRAYS * 32;  // NRAYS
    float* out_wsm = out_dep + NRAYS;           // NRAYS
    float* out_xyz = out_wsm + NRAYS;           // NRAYS*3
    for (int k = 0; k < 32; ++k) out_rgb[(size_t)ray * 32 + k] = comp[k] * 2.f - 1.f;
    out_dep[ray] = depth;
    out_wsm[ray] = wsum;
    for (int k = 0; k < 3; ++k) out_xyz[(size_t)ray * 3 + k] = comp[32 + k] * 2.f - 1.f;
}

// ---------------------------------------------------------------------------
extern "C" void kernel_launch(void* const* d_in, const int* in_sizes, int n_in,
                              void* d_out, int out_size, void* d_ws, size_t ws_size,
                              hipStream_t stream) {
    const float* front  = (const float*)d_in[0];
    const float* back   = (const float*)d_in[1];
    const float* ro     = (const float*)d_in[2];
    const float* rd     = (const float*)d_in[3];
    const float* weight = (const float*)d_in[4];
    const float* w1     = (const float*)d_in[5];
    const float* b1     = (const float*)d_in[6];
    const float* w2     = (const float*)d_in[7];
    const float* b2     = (const float*)d_in[8];
    float* out = (float*)d_out;

    float* ws = (float*)d_ws;
    float* depthsC = ws;
    float* depthsF = depthsC + MC;
    float* sigmaC  = depthsF + MC;
    float* sigmaF  = sigmaC + MC;
    float* colorC  = sigmaF + MC;
    float* colorF  = colorC + (size_t)MC * 32;

    gen_depths<<<(MC + 255) / 256, 256, 0, stream>>>(depthsC);

    // one wave per 16-sample tile, 8 waves per block
    const int blocks = MC / (16 * 8);  // 3072
    fused_sample_mlp<<<blocks, 256, 0, stream>>>(
        front, back, ro, rd, weight, w1, b1, w2, b2,
        depthsC, colorC, sigmaC);

    importance_sample<<<(NRAYS + 255) / 256, 256, 0, stream>>>(
        depthsC, sigmaC, depthsF);

    fused_sample_mlp<<<blocks, 256, 0, stream>>>(
        front, back, ro, rd, weight, w1, b1, w2, b2,
        depthsF, colorF, sigmaF);

    final_march<<<(NRAYS + 255) / 256, 256, 0, stream>>>(
        depthsC, depthsF, sigmaC, sigmaF, colorC, colorF, ro, rd, out);
}
